// SPMoEAdaptor_26680336843012
// MI455X (gfx1250) — matrix operand order repaired and checked by reading, not compiled, not run on hardware
//
#include <hip/hip_runtime.h>

// ---------------------------------------------------------------------------
// Fused 2-layer soft-MoE for MI455X (gfx1250, wave32, WMMA).
//   out = moe_b(moe_a(x)) + x
//   moe(x) = sum_e softmax(x@wg)[:,e] * ((x - b_e) @ W_e)
//          = sum_e g_e*(x@W_e) - g @ C,   C[e,:] = b_e @ W_e
// Transposed WMMA mapping: D = W^T (A, LDS-resident) x x^T (B, streamed).
// HBM-bound target: read x once (134 MB), write out once (134 MB).
// ---------------------------------------------------------------------------

typedef _Float16 v16h __attribute__((ext_vector_type(16)));  // WMMA A/B operand
typedef __fp16   h2   __attribute__((ext_vector_type(2)));   // cvt_pkrtz result
typedef float    v8f  __attribute__((ext_vector_type(8)));
typedef float    f4   __attribute__((ext_vector_type(4)));

union V16u { v16h v; h2 p[8]; };

__device__ __forceinline__ float h2f(h2 x) { union { h2 h; float f; } u; u.h = x; return u.f; }
__device__ __forceinline__ h2 f2h(float x) { union { h2 h; float f; } u; u.f = x; return u.h; }
__device__ __forceinline__ v16h ldA(const _Float16* p) { return *(const v16h*)p; }

#define WMMA(a, b, c) \
  __builtin_amdgcn_wmma_f32_16x16x32_f16(false, (a), false, (b), (short)0, (c), false, false)

// One MoE layer for two concurrent 16-token subtiles.
// B[s][kb]  : f16 B-operand tiles of the layer input (token-major in lanes)
// acc[s][fb]: f32 accumulators, D layout (token in lane, feature in VGPR)
__device__ __forceinline__ void moe_layer(
    const _Float16* __restrict__ sW,   // 32 pre-swizzled A tiles (e,fb,kb)
    const _Float16* __restrict__ sG,   // 2 gate A tiles (kb), rows padded to 16
    const _Float16* __restrict__ sC,   // 4 bias-correction A tiles (fb), K padded
    int lane, int hi, const V16u B[2][2], v8f acc[2][4]) {
  // ---- gate logits: D[m=e][n=t] -------------------------------------------
  v8f ga[2] = {{}, {}};
#pragma unroll
  for (int kb = 0; kb < 2; ++kb) {
    v16h a = ldA(sG + kb * 512 + lane * 16);
    ga[0] = WMMA(a, B[0][kb].v, ga[0]);
    ga[1] = WMMA(a, B[1][kb].v, ga[1]);
  }
  // ---- per-lane softmax over e=0..3 (expert index is the VGPR dim) --------
  float g[2][4];
#pragma unroll
  for (int s = 0; s < 2; ++s) {
    float l0 = ga[s][0], l1 = ga[s][1], l2 = ga[s][2], l3 = ga[s][3];
    float mx = fmaxf(fmaxf(l0, l1), fmaxf(l2, l3));
    float e0 = __expf(l0 - mx), e1 = __expf(l1 - mx);
    float e2 = __expf(l2 - mx), e3 = __expf(l3 - mx);
    float inv = 1.0f / (e0 + e1 + e2 + e3);
    g[s][0] = e0 * inv; g[s][1] = e1 * inv; g[s][2] = e2 * inv; g[s][3] = e3 * inv;
    // valid gates live in lanes 0..15 (token t = lane); broadcast to lane t+16
#pragma unroll
    for (int e = 0; e < 4; ++e) g[s][e] = __shfl(g[s][e], lane & 15, 32);
  }
  // ---- bias correction: D += (-C)^T x g^T  (K=4, zero padded) -------------
  V16u Bg[2];
#pragma unroll
  for (int s = 0; s < 2; ++s) {
    h2 z = {};
#pragma unroll
    for (int q = 0; q < 8; ++q) Bg[s].p[q] = z;
    if (!hi) {  // B nonzero only at k=0..3 (lo lanes hold k=0..15)
      Bg[s].p[0] = __builtin_amdgcn_cvt_pkrtz(g[s][0], g[s][1]);
      Bg[s].p[1] = __builtin_amdgcn_cvt_pkrtz(g[s][2], g[s][3]);
    }
  }
#pragma unroll
  for (int fb = 0; fb < 4; ++fb) {
    v16h a = ldA(sC + fb * 512 + lane * 16);
    acc[0][fb] = WMMA(a, Bg[0].v, acc[0][fb]);
    acc[1][fb] = WMMA(a, Bg[1].v, acc[1][fb]);
  }
  // ---- experts: gate folded into B operand, C-chain accumulation ----------
  // kb outer / fb inner keeps only one scaled-B set (16 VGPRs) live.
#pragma unroll
  for (int e = 0; e < 4; ++e) {
    h2 gh[2];
    gh[0] = __builtin_amdgcn_cvt_pkrtz(g[0][e], g[0][e]);
    gh[1] = __builtin_amdgcn_cvt_pkrtz(g[1][e], g[1][e]);
#pragma unroll
    for (int kb = 0; kb < 2; ++kb) {
      V16u Be[2];
#pragma unroll
      for (int s = 0; s < 2; ++s)
#pragma unroll
        for (int q = 0; q < 8; ++q) Be[s].p[q] = B[s][kb].p[q] * gh[s];
#pragma unroll
      for (int fb = 0; fb < 4; ++fb) {
        v16h a = ldA(sW + ((e * 4 + fb) * 2 + kb) * 512 + lane * 16);
        acc[0][fb] = WMMA(a, Be[0].v, acc[0][fb]);
        acc[1][fb] = WMMA(a, Be[1].v, acc[1][fb]);
      }
    }
  }
}

__global__ void __launch_bounds__(256) moe2_kernel(
    const float* __restrict__ x,
    const float* __restrict__ wg_a, const float* __restrict__ we_a,
    const float* __restrict__ be_a,
    const float* __restrict__ wg_b, const float* __restrict__ we_b,
    const float* __restrict__ be_b,
    float* __restrict__ out, int N, int chunks) {
  extern __shared__ _Float16 smem[];
  _Float16* sW = smem;            // 2 layers x 16384 halves (A-swizzled W_e)
  _Float16* sG = smem + 32768;    // 2 layers x 2 kb x 512  (gate A tiles)
  _Float16* sC = smem + 34816;    // 2 layers x 4 fb x 512  (-C A tiles, K=4 pad)

  const int tid = threadIdx.x;
  // -------- one-time staging: swizzle f32 weights -> f16 A-operand layout --
  for (int i = tid; i < 6144; i += 256) sG[i] = (_Float16)0.0f;  // sG + sC
  for (int i = tid; i < 32768; i += 256) {
    int layer = i >> 14, r = i & 16383;
    int tile = r >> 9, l = (r >> 4) & 31, j = r & 15;
    int kb = tile & 1, fb = (tile >> 1) & 3, e = tile >> 3;
    int hi = l >> 4, m = l & 15;
    int k = kb * 32 + hi * 16 + j;
    int f = fb * 16 + m;
    const float* W = layer ? we_b : we_a;
    sW[i] = (_Float16)W[(e * 64 + k) * 64 + f];
  }
  for (int i = tid; i < 2048; i += 256) {
    int layer = i >> 10, r = i & 1023;
    int kb = r >> 9, l = (r >> 4) & 31, j = r & 15;
    int hi = l >> 4, m = l & 15;
    int k = kb * 32 + hi * 16 + j;
    const float* G = layer ? wg_b : wg_a;
    sG[i] = (m < 4) ? (_Float16)G[k * 4 + m] : (_Float16)0.0f;
  }
  for (int i = tid; i < 512; i += 256) {  // C[e,f] = b_e @ W_e (negated)
    int layer = i >> 8, r = i & 255;
    int e = r >> 6, fb = (r >> 4) & 3, m = r & 15;
    int f = fb * 16 + m;
    const float* W = layer ? we_b : we_a;
    const float* B = layer ? be_b : be_a;
    float s = 0.0f;
    for (int d = 0; d < 64; ++d) s += B[e * 64 + d] * W[(e * 64 + d) * 64 + f];
    sC[(layer * 4 + fb) * 512 + m * 16 + e] = (_Float16)(-s);  // lane m, k=e
  }
  __syncthreads();

  const int lane = tid & 31, wv = tid >> 5;
  const int hi = lane >> 4, m = lane & 15;

  for (int c = blockIdx.x; c < chunks; c += gridDim.x) {
    // Opaque barrier: forbid hoisting the (loop-invariant) LDS A-tile loads
    // out of this loop — last round LICM hoisted ~600 VGPRs of A tiles and
    // spilled them to scratch. LDS is the intended fast path.
    asm volatile("" ::: "memory");

    const int tbase = c * 256 + wv * 32;  // this wave: 2 subtiles of 16 tokens
    // ---- stream x: lane holds x[t, kb*32 + hi*16 + (0..15)] in f32 --------
    float xf[2][2][16];
#pragma unroll
    for (int s = 0; s < 2; ++s) {
      int row = tbase + s * 16 + m;
      int rowl = row < N ? row : N - 1;
      const float* xp = x + (size_t)rowl * 64 + hi * 16;
#pragma unroll
      for (int kb = 0; kb < 2; ++kb)
#pragma unroll
        for (int q = 0; q < 4; ++q) {
          f4 t = __builtin_nontemporal_load((const f4*)(xp + kb * 32 + q * 4));
#pragma unroll
          for (int i = 0; i < 4; ++i) xf[s][kb][q * 4 + i] = t[i];
        }
    }
    V16u B1[2][2];
#pragma unroll
    for (int s = 0; s < 2; ++s)
#pragma unroll
      for (int kb = 0; kb < 2; ++kb)
#pragma unroll
        for (int q = 0; q < 8; ++q)
          B1[s][kb].p[q] =
              __builtin_amdgcn_cvt_pkrtz(xf[s][kb][2 * q], xf[s][kb][2 * q + 1]);

    // ---- layer A -----------------------------------------------------------
    v8f acc1[2][4] = {};
    moe_layer(sW, sG, sC, lane, hi, B1, acc1);

    // ---- repack h (D layout) -> layer-B B operand: cvt + xor16 exchange ----
    V16u B2[2][2];
#pragma unroll
    for (int s = 0; s < 2; ++s) {
      float pkf[4][4];
#pragma unroll
      for (int fb = 0; fb < 4; ++fb)
#pragma unroll
        for (int v = 0; v < 4; ++v)
          pkf[fb][v] = h2f(__builtin_amdgcn_cvt_pkrtz(acc1[s][fb][2 * v],
                                                      acc1[s][fb][2 * v + 1]));
#pragma unroll
      for (int kb = 0; kb < 2; ++kb) {
        float ex[4];
#pragma unroll
        for (int v = 0; v < 4; ++v) {
          float snd = hi ? pkf[2 * kb][v] : pkf[2 * kb + 1][v];  // sender flip
          ex[v] = __shfl_xor(snd, 16, 32);
        }
#pragma unroll
        for (int v = 0; v < 4; ++v) {
          B2[s][kb].p[v]     = f2h(hi ? ex[v] : pkf[2 * kb][v]);
          B2[s][kb].p[4 + v] = f2h(hi ? pkf[2 * kb + 1][v] : ex[v]);
        }
      }
    }

    // ---- layer B -----------------------------------------------------------
    v8f acc2[2][4] = {};
    moe_layer(sW + 16384, sG + 1024, sC + 2048, lane, hi, B2, acc2);

    // ---- residual (register-resident x, one xor16 exchange) + store --------
#pragma unroll
    for (int s = 0; s < 2; ++s) {
      int row = tbase + s * 16 + m;  // uniform across xor-16 partners
      if (row < N) {
        float sx[2][8];
#pragma unroll
        for (int a = 0; a < 2; ++a)
#pragma unroll
          for (int r = 0; r < 8; ++r) {
            float snd = hi ? xf[s][a][r] : xf[s][a][8 + r];  // sender flip
            sx[a][r] = __shfl_xor(snd, 16, 32);
          }
        float* op = out + (size_t)row * 64 + hi * 8;
#pragma unroll
        for (int fb = 0; fb < 4; ++fb) {
          int a = fb >> 1;
          f4 o0, o1;
#pragma unroll
          for (int r = 0; r < 8; ++r) {
            float own = hi ? xf[s][a][8 + r] : xf[s][a][r];
            float res = ((fb & 1) == hi) ? own : sx[a][r];
            float v = acc2[s][fb][r] + res;
            if (r < 4) o0[r] = v; else o1[r - 4] = v;
          }
          __builtin_nontemporal_store(o0, (f4*)(op + fb * 16));
          __builtin_nontemporal_store(o1, (f4*)(op + fb * 16 + 4));
        }
      }
    }
  }
}

extern "C" void kernel_launch(void* const* d_in, const int* in_sizes, int n_in,
                              void* d_out, int out_size, void* d_ws, size_t ws_size,
                              hipStream_t stream) {
  const float* x    = (const float*)d_in[0];
  const float* wg_a = (const float*)d_in[1];
  const float* we_a = (const float*)d_in[2];
  const float* be_a = (const float*)d_in[3];
  const float* wg_b = (const float*)d_in[4];
  const float* we_b = (const float*)d_in[5];
  const float* be_b = (const float*)d_in[6];
  float* out = (float*)d_out;

  int N = in_sizes[0] / 64;
  int chunks = (N + 255) / 256;
  int grid = chunks < 512 ? chunks : 512;
  if (grid < 1) grid = 1;
  size_t smem = 38912 * sizeof(_Float16);  // 76 KB: W(64K) + gate(4K) + C(8K)

  (void)hipFuncSetAttribute(reinterpret_cast<const void*>(moe2_kernel),
                            hipFuncAttributeMaxDynamicSharedMemorySize, (int)smem);
  moe2_kernel<<<grid, 256, smem, stream>>>(x, wg_a, we_a, be_a, wg_b, we_b, be_b,
                                           out, N, chunks);
}
